// M_2Simp_AB_87273735454745
// MI455X (gfx1250) — matrix operand order repaired and checked by reading, not compiled, not run on hardware
//
#include <hip/hip_runtime.h>

// ---------------------------------------------------------------------------
// 2-simplicial attention for MI455X (gfx1250, wave32, WMMA).
//   B=2, N=160, D=512, H=8, HD=64.
// Flash-style fusion over the (k,l) product space: never materialize the
// 262MB bhqkl tensor (HBM-bound at 23.3 TB/s if materialized); keep per-(b,h)
// K/V/K2/V2 in LDS (320KB/WGP) and stream l. Matrix work on
// v_wmma_f32_16x16x32_f16 (f16 operands, f32 accum); softmax/norm in f32.
// All fragment traffic is 128-bit (global_load_b128 / ds_load_b128).
// ---------------------------------------------------------------------------

typedef __attribute__((ext_vector_type(16))) _Float16 v16h;
typedef __attribute__((ext_vector_type(8)))  _Float16 v8h;
typedef __attribute__((ext_vector_type(8)))  float    v8f;
typedef __attribute__((ext_vector_type(4)))  float    f4;

union H16 { v16h v; v8h h[2]; };
union F4  { f4 v; float f[4]; };

#define Bx 2
#define Nx 160
#define Dx 512
#define Hx 8
#define HDx 64
#define MROWS (Bx*Nx)     // 320

// LDS row strides (f16 elements). Byte strides 144/336 keep 16B alignment and
// gcd(dword_stride,64) <= 4 so 16-lane column walks hit distinct banks.
#define KS   72           // k_s  rows: [n][d]
#define VTS  168          // v_t  rows: [d][n]  (transposed V)
#define PSTR 168          // P staging slab rows

// CDNA5 ISA 7.12.2, 16-bit A-fragment (16x32): element e of the v16h holds
//   K = 8*half + e          for e in [0,8)    (VGPRs 0..3)
//   K = 16 + 8*half + (e-8) for e in [8,16)   (VGPRs 4..7)
// i.e. two contiguous 8-element runs -> two 16B loads.
// 16-bit B-fragment (32x16): same K order as A (two 8-runs -> two 16B loads).

__device__ __forceinline__ v8f wmma_f16(v16h a, v16h b, v8f c) {
  // 8 args: (neg_a, A, neg_b, B, c_mod, C, reuse_a, reuse_b)
  return __builtin_amdgcn_wmma_f32_16x16x32_f16(false, a, false, b, (short)0, c,
                                                false, false);
}

// ---------------------------------------------------------------------------
// Projection GEMM: out[m,n] = (sum_k X[m,k]*W[n,k] + bias[n]) * scale
// One wave per 16x16 tile; 16 K-chunks of 32 -> 16 WMMAs; all loads b128.
// ---------------------------------------------------------------------------
__global__ void proj_kernel(const float* __restrict__ X, const float* __restrict__ W,
                            const float* __restrict__ bias, float scale,
                            _Float16* __restrict__ outH, float* __restrict__ outHf,
                            float* __restrict__ outRow) {
  const int lane = threadIdx.x & 31;
  const int wid  = threadIdx.x >> 5;
  const int lm = lane & 15, half = lane >> 4;
  const int m0 = blockIdx.x * 16;
  const int n0 = (blockIdx.y * 4 + wid) * 16;

  const f4* xr = (const f4*)(X + (size_t)(m0 + lm) * Dx);
  const f4* wr = (const f4*)(W + (size_t)(n0 + lm) * Dx);

  v8f acc = {};
  for (int c4 = 0; c4 < Dx / 4; c4 += 8) {            // c4*4 = K-chunk base
    if (c4 + 16 < Dx / 4)                             // CDNA5 global_prefetch_b8
      __builtin_prefetch(&wr[c4 + 16], 0, 0);
    F4 ax[4], bx[4];
    ax[0].v = xr[c4 + 2 * half];     ax[1].v = xr[c4 + 2 * half + 1];
    ax[2].v = xr[c4 + 4 + 2 * half]; ax[3].v = xr[c4 + 4 + 2 * half + 1];
    bx[0].v = wr[c4 + 2 * half];     bx[1].v = wr[c4 + 2 * half + 1];
    bx[2].v = wr[c4 + 4 + 2 * half]; bx[3].v = wr[c4 + 4 + 2 * half + 1];
    v16h a, b;
#pragma unroll
    for (int e = 0; e < 8; ++e) {
      a[e]     = (_Float16)ax[e >> 2].f[e & 3];
      a[e + 8] = (_Float16)ax[2 + (e >> 2)].f[e & 3];
      b[e]     = (_Float16)bx[e >> 2].f[e & 3];
      b[e + 8] = (_Float16)bx[2 + (e >> 2)].f[e & 3];
    }
    acc = wmma_f16(a, b, acc);
  }

  const int ncol = n0 + lm;
  const float bn = bias[ncol];
#pragma unroll
  for (int r = 0; r < 8; ++r) {
    const int M = m0 + r + 8 * half;                  // C/D: VGPR r -> row r / r+8
    const float val = (acc[r] + bn) * scale;
    if (outRow) outRow[(size_t)M * Dx + ncol] = val;
    const int bidx = M / Nx, n = M % Nx;
    const int h = ncol >> 6, d = ncol & 63;
    const size_t hi = (((size_t)(bidx * Hx + h)) * Nx + n) * HDx + d;
    if (outH)  outH[hi]  = (_Float16)val;
    if (outHf) outHf[hi] = val;
  }
}

// ---------------------------------------------------------------------------
// Fused 2-simplicial flash attention.
// grid = B*H blocks, block = 320 threads = 10 wave32s; wave w owns q rows
// [16w, 16w+16). For each l in 0..159:
//   S = (q .* k2[l]) @ k^T          (10x 2-chained WMMAs)
//   online softmax update (row max/sum via 16-lane shfl reductions)
//   acc += (P @ v) .* v2[l]         (4x  5-chained WMMAs; A-frags hoisted)
// P goes D-layout -> LDS -> A-layout (same-wave DS ops are in-order).
// ---------------------------------------------------------------------------
__global__ void attn2s_kernel(const _Float16* __restrict__ qh,
                              const _Float16* __restrict__ kh,
                              const _Float16* __restrict__ vh,
                              const float* __restrict__ k2h,
                              const float* __restrict__ v2h,
                              float* __restrict__ xo) {
  extern __shared__ char smem[];
  _Float16* k_s  = (_Float16*)smem;                   // [160][KS]  f16
  _Float16* v_t  = k_s + Nx * KS;                     // [64][VTS]  f16 (V^T)
  float*    k2_s = (float*)(v_t + HDx * VTS);         // [160][64]  f32
  float*    v2_s = k2_s + Nx * HDx;                   // [160][64]  f32
  _Float16* p_s  = (_Float16*)(v2_s + Nx * HDx);      // 10*[16][PSTR] f16

  const int bh = blockIdx.x;
  const int tid = threadIdx.x;
  const int wave = tid >> 5, lane = tid & 31, lm = lane & 15, half = lane >> 4;

  const size_t boff = (size_t)bh * Nx * HDx;
  const _Float16* qg = qh + boff;
  const _Float16* kg = kh + boff;
  const _Float16* vg = vh + boff;
  const float* k2g = k2h + boff;
  const float* v2g = v2h + boff;

  // cooperative fills: k row-major (16B chunks), v transposed, k2/v2 f32x4
  for (int i = tid; i < Nx * HDx / 8; i += blockDim.x) {
    const int n = i >> 3, c = i & 7;
    *(v8h*)(k_s + n * KS + c * 8) = *(const v8h*)(kg + n * HDx + c * 8);
  }
  for (int i = tid; i < Nx * HDx; i += blockDim.x) {
    const int n = i >> 6, d = i & 63;
    v_t[d * VTS + n] = vg[i];
  }
  for (int i = tid; i < Nx * HDx / 4; i += blockDim.x) {
    ((f4*)k2_s)[i] = ((const f4*)k2g)[i];
    ((f4*)v2_s)[i] = ((const f4*)v2g)[i];
  }
  __syncthreads();

  const int m0 = wave * 16;

  // q rows fp32, A-fragment order: two 8-runs per 32-chunk (b128 loads)
  const v8h* qp = (const v8h*)(qg + (size_t)(m0 + lm) * HDx);
  float qreg[2][16];
#pragma unroll
  for (int c = 0; c < 2; ++c) {
    const v8h r0 = qp[4 * c + half], r1 = qp[4 * c + 2 + half];
#pragma unroll
    for (int e = 0; e < 8; ++e) {
      qreg[c][e]     = (float)r0[e];
      qreg[c][e + 8] = (float)r1[e];
    }
  }

  float m_run[8], l_run[8];
#pragma unroll
  for (int r = 0; r < 8; ++r) { m_run[r] = -3.0e38f; l_run[r] = 0.f; }
  v8f acc[4];
#pragma unroll
  for (int t = 0; t < 4; ++t) acc[t] = (v8f){};

  _Float16* pw = p_s + wave * 16 * PSTR;

  for (int l = 0; l < Nx; ++l) {
    // ---- k2[l] in fragment order (4x ds_load_b128 per chunk pair) ----
    const f4* k2p = (const f4*)(k2_s + l * HDx);
    float k2v[2][16];
#pragma unroll
    for (int c = 0; c < 2; ++c) {
      F4 r0, r1, r2, r3;
      r0.v = k2p[8 * c + 2 * half];     r1.v = k2p[8 * c + 2 * half + 1];
      r2.v = k2p[8 * c + 4 + 2 * half]; r3.v = k2p[8 * c + 4 + 2 * half + 1];
#pragma unroll
      for (int e = 0; e < 4; ++e) {
        k2v[c][e]      = r0.f[e];
        k2v[c][e + 4]  = r1.f[e];
        k2v[c][e + 8]  = r2.f[e];
        k2v[c][e + 12] = r3.f[e];
      }
    }
    // ---- A fragments: q .* k2[l] (f32 multiply, then f16 cast) ----
    v16h a0, a1;
#pragma unroll
    for (int e = 0; e < 16; ++e) {
      a0[e] = (_Float16)(qreg[0][e] * k2v[0][e]);
      a1[e] = (_Float16)(qreg[1][e] * k2v[1][e]);
    }

    // ---- S = A @ k^T : 10 column tiles, K=64 as 2 chained WMMAs ----
    v8f s[10];
#pragma unroll
    for (int j = 0; j < 10; ++j) {
      const v8h* kb = (const v8h*)(k_s + (16 * j + lm) * KS + 8 * half);
      H16 b0, b1;
      b0.h[0] = kb[0]; b0.h[1] = kb[2];   // K = 8*half + 0..7, 16 + 8*half + 0..7
      b1.h[0] = kb[4]; b1.h[1] = kb[6];   // K = 32 + same
      v8f z = {};
      z = wmma_f16(a0, b0.v, z);
      s[j] = wmma_f16(a1, b1.v, z);
    }

    // ---- online softmax across this l-slab (rows live in 16-lane halves) ----
#pragma unroll
    for (int r = 0; r < 8; ++r) {
      float mt = s[0][r];
#pragma unroll
      for (int j = 1; j < 10; ++j) mt = fmaxf(mt, s[j][r]);
#pragma unroll
      for (int off = 8; off >= 1; off >>= 1)
        mt = fmaxf(mt, __shfl_xor(mt, off, 16));
      const float mn = fmaxf(m_run[r], mt);
      const float cf = __expf(m_run[r] - mn);
      m_run[r] = mn;

      float rs = 0.f;
#pragma unroll
      for (int j = 0; j < 10; ++j) {
        const float p = __expf(s[j][r] - mn);
        s[j][r] = p;
        rs += p;
      }
#pragma unroll
      for (int off = 8; off >= 1; off >>= 1)
        rs += __shfl_xor(rs, off, 16);
      l_run[r] = l_run[r] * cf + rs;

#pragma unroll
      for (int t = 0; t < 4; ++t) acc[t][r] *= cf;

      // stage P: D-layout (col=16j+lm, row=r+8*half) -> row-major LDS
#pragma unroll
      for (int j = 0; j < 10; ++j)
        pw[(r + 8 * half) * PSTR + 16 * j + lm] = (_Float16)s[j][r];
    }

    // ---- T = P @ v ; acc += T .* v2[l] ----
    // A fragments depend only on the K-chunk: build once (b128 loads)
    v16h af[5];
#pragma unroll
    for (int c = 0; c < 5; ++c) {
      const v8h* pp = (const v8h*)(pw + lm * PSTR + 32 * c + 8 * half);
      H16 au;
      au.h[0] = pp[0];       // K = 32c + 8*half + 0..7
      au.h[1] = pp[2];       // K = 32c + 16 + 8*half + 0..7
      af[c] = au.v;
    }
#pragma unroll
    for (int t = 0; t < 4; ++t) {
      v8f z = {};
#pragma unroll
      for (int c = 0; c < 5; ++c) {
        const v8h* vb = (const v8h*)(v_t + (16 * t + lm) * VTS + 32 * c + 8 * half);
        H16 bu;
        bu.h[0] = vb[0]; bu.h[1] = vb[2];   // K = 32c + 8*half + 0..7, 32c + 16 + 8*half + 0..7
        z = wmma_f16(af[c], bu.v, z);
      }
      const float v2l = v2_s[l * HDx + 16 * t + lm];
#pragma unroll
      for (int r = 0; r < 8; ++r) acc[t][r] += z[r] * v2l;
    }
  }

  // normalize and write [b,h,q,d] (== the reference's no-transpose reshape)
#pragma unroll
  for (int t = 0; t < 4; ++t)
#pragma unroll
    for (int r = 0; r < 8; ++r) {
      const int q = m0 + r + 8 * half;
      xo[((size_t)bh * Nx + q) * HDx + 16 * t + lm] = acc[t][r] / l_run[r];
    }
}

// ---------------------------------------------------------------------------
// Epilogue: out = X + mish(X @ Wo^T + bo),  mish(y) = y * tanh(softplus(y))
// ---------------------------------------------------------------------------
__global__ void out_mish_kernel(const float* __restrict__ X, const float* __restrict__ W,
                                const float* __restrict__ bias,
                                float* __restrict__ out) {
  const int lane = threadIdx.x & 31;
  const int wid  = threadIdx.x >> 5;
  const int lm = lane & 15, half = lane >> 4;
  const int m0 = blockIdx.x * 16;
  const int n0 = (blockIdx.y * 4 + wid) * 16;

  const f4* xr = (const f4*)(X + (size_t)(m0 + lm) * Dx);
  const f4* wr = (const f4*)(W + (size_t)(n0 + lm) * Dx);

  v8f acc = {};
  for (int c4 = 0; c4 < Dx / 4; c4 += 8) {
    F4 ax[4], bx[4];
    ax[0].v = xr[c4 + 2 * half];     ax[1].v = xr[c4 + 2 * half + 1];
    ax[2].v = xr[c4 + 4 + 2 * half]; ax[3].v = xr[c4 + 4 + 2 * half + 1];
    bx[0].v = wr[c4 + 2 * half];     bx[1].v = wr[c4 + 2 * half + 1];
    bx[2].v = wr[c4 + 4 + 2 * half]; bx[3].v = wr[c4 + 4 + 2 * half + 1];
    v16h a, b;
#pragma unroll
    for (int e = 0; e < 8; ++e) {
      a[e]     = (_Float16)ax[e >> 2].f[e & 3];
      a[e + 8] = (_Float16)ax[2 + (e >> 2)].f[e & 3];
      b[e]     = (_Float16)bx[e >> 2].f[e & 3];
      b[e + 8] = (_Float16)bx[2 + (e >> 2)].f[e & 3];
    }
    acc = wmma_f16(a, b, acc);
  }

  const int ncol = n0 + lm;
  const float bn = bias[ncol];
#pragma unroll
  for (int r = 0; r < 8; ++r) {
    const int M = m0 + r + 8 * half;
    const float y = acc[r] + bn;
    const float sp = (y > 20.f) ? y : log1pf(__expf(y));
    const float mish = y * tanhf(sp);
    out[(size_t)M * Dx + ncol] = X[(size_t)M * Dx + ncol] + mish;
  }
}

// ---------------------------------------------------------------------------
extern "C" void kernel_launch(void* const* d_in, const int* in_sizes, int n_in,
                              void* d_out, int out_size, void* d_ws, size_t ws_size,
                              hipStream_t stream) {
  (void)in_sizes; (void)n_in; (void)out_size; (void)ws_size;
  const float* Q   = (const float*)d_in[0];
  const float* K   = (const float*)d_in[1];
  const float* Wq  = (const float*)d_in[2];  const float* bq  = (const float*)d_in[3];
  const float* Wk  = (const float*)d_in[4];  const float* bk  = (const float*)d_in[5];
  const float* Wv  = (const float*)d_in[6];  const float* bv  = (const float*)d_in[7];
  const float* Wk2 = (const float*)d_in[8];  const float* bk2 = (const float*)d_in[9];
  const float* Wv2 = (const float*)d_in[10]; const float* bv2 = (const float*)d_in[11];
  const float* Wo  = (const float*)d_in[12]; const float* bo  = (const float*)d_in[13];

  // workspace: head-layout f16 q/k/v, head-layout f32 k2/v2, row-major f32 Kp,
  // attention output f32. Total ~3.44 MB.
  char* ws = (char*)d_ws;
  _Float16* q_h  = (_Float16*)(ws + 0);        // 327680 B
  _Float16* k_h  = (_Float16*)(ws + 327680);   // 327680 B
  _Float16* v_h  = (_Float16*)(ws + 655360);   // 327680 B
  float*    k2_h = (float*)(ws + 983040);      // 655360 B
  float*    v2_h = (float*)(ws + 1638400);     // 655360 B
  float*    kp   = (float*)(ws + 2293760);     // 655360 B
  float*    xo   = (float*)(ws + 2949120);     // 655360 B

  dim3 gp(MROWS / 16, Dx / 16 / 4), bp(128);
  const float qscale = 0.125f;  // HD^-0.5, folded into Qp (uniform row scale)

  // stage 1: Qp (scaled), Kp (+ f32 copy for stage 2), V (from raw K)
  proj_kernel<<<gp, bp, 0, stream>>>(Q, Wq, bq, qscale, q_h, nullptr, nullptr);
  proj_kernel<<<gp, bp, 0, stream>>>(K, Wk, bk, 1.0f, k_h, nullptr, kp);
  proj_kernel<<<gp, bp, 0, stream>>>(K, Wv, bv, 1.0f, v_h, nullptr, nullptr);
  // stage 2: K2/V2 from projected Kp (stream order gives the dependency)
  proj_kernel<<<gp, bp, 0, stream>>>(kp, Wk2, bk2, 1.0f, nullptr, k2_h, nullptr);
  proj_kernel<<<gp, bp, 0, stream>>>(kp, Wv2, bv2, 1.0f, nullptr, v2_h, nullptr);

  const size_t lds = (size_t)(Nx * KS * 2)         // k_s      = 23040
                   + (size_t)(HDx * VTS * 2)       // v_t      = 21504
                   + (size_t)(2 * Nx * HDx * 4)    // k2_s,v2_s= 81920
                   + (size_t)(10 * 16 * PSTR * 2); // P slab   = 53760 -> 180224 B
  attn2s_kernel<<<dim3(Bx * Hx), dim3(320), lds, stream>>>(q_h, k_h, v_h, k2_h,
                                                           v2_h, xo);

  out_mish_kernel<<<gp, bp, 0, stream>>>(xo, Wo, bo, (float*)d_out);
}
